// SOT_69037304316173
// MI455X (gfx1250) — matrix operand order repaired and checked by reading, not compiled, not run on hardware
//
#include <hip/hip_runtime.h>
#include <math.h>

// ---------------------------------------------------------------------------
// SOT forward for MI455X (gfx1250, wave32, WMMA).
//   N = 8192, D = 128, OT_REG = 0.1, ITERS = 10, DIAG = 1000.
// Strategy: materialize log_K into d_out (exactly N*N floats) with a
// split-f16 WMMA GEMM, then bandwidth-bound streaming LSE passes.
// ---------------------------------------------------------------------------

static constexpr int kN   = 8192;
static constexpr int kD   = 128;
static constexpr int kRC  = 16;            // row chunks for column-LSE pass
static constexpr int kRCH = kN / kRC;      // 512 rows per chunk

typedef __attribute__((ext_vector_type(16))) _Float16 v16h;
typedef __attribute__((ext_vector_type(8)))  _Float16 v8h;
typedef __attribute__((ext_vector_type(8)))  float    v8f;

// ---------------------------------------------------------------------------
// 1) Row-normalize X and emit split-precision f16 (hi + lo == fp32 value).
//    One wave per row (32 lanes x 4 elems), wave32 shuffle reduction.
// ---------------------------------------------------------------------------
__global__ void sot_normalize(const float* __restrict__ X,
                              _Float16* __restrict__ hi,
                              _Float16* __restrict__ lo) {
  const int wave = threadIdx.x >> 5;
  const int lane = threadIdx.x & 31;
  const int row  = blockIdx.x * 8 + wave;          // 256 threads = 8 rows
  const float* xr = X + (size_t)row * kD;

  float x[4];
#pragma unroll
  for (int k = 0; k < 4; ++k) x[k] = xr[lane + 32 * k];

  float ss = x[0] * x[0] + x[1] * x[1] + x[2] * x[2] + x[3] * x[3];
#pragma unroll
  for (int m = 16; m >= 1; m >>= 1) ss += __shfl_xor(ss, m, 32);

  const float rn = 1.0f / sqrtf(ss);
#pragma unroll
  for (int k = 0; k < 4; ++k) {
    const float    xn = x[k] * rn;
    const _Float16 h  = (_Float16)xn;
    const _Float16 l  = (_Float16)(xn - (float)h);
    hi[(size_t)row * kD + lane + 32 * k] = h;
    lo[(size_t)row * kD + lane + 32 * k] = l;
  }
}

// ---------------------------------------------------------------------------
// 2) log_K = 10*(Xn.Xn^T - 1), diag = -10000, via v_wmma_f32_16x16x32_f16.
//    Split-precision: C ~= hi.hi + hi.lo + lo.hi  (3 independent accums).
//    Block = 256 threads = 8 waves; wave w owns the 16x16 tile
//    (blockIdx.x*16, blockIdx.y*128 + 16*w).
// ---------------------------------------------------------------------------
__global__ void sot_logk(const _Float16* __restrict__ Ahi,
                         const _Float16* __restrict__ Alo,
                         float* __restrict__ logK) {
  const int lane  = threadIdx.x & 31;
  const int wave  = threadIdx.x >> 5;
  const int halfw = lane >> 4;       // which 16-lane half
  const int idx16 = lane & 15;
  const int tm    = blockIdx.x * 16;
  const int tn    = blockIdx.y * 128 + wave * 16;

  // A fragment: lane (halfw, m=idx16) holds row tm+idx16,
  //   K = kc + halfw*8 + [0..7]  and  kc + 16 + halfw*8 + [0..7].
  // B fragment (= Xn^T): lane (halfw, n=idx16) holds row tn+idx16,
  //   K = kc + halfw*8 + [0..7]  and  kc + 16 + halfw*8 + [0..7].
  const _Float16* arh = Ahi + (size_t)(tm + idx16) * kD;
  const _Float16* arl = Alo + (size_t)(tm + idx16) * kD;
  const _Float16* brh = Ahi + (size_t)(tn + idx16) * kD;
  const _Float16* brl = Alo + (size_t)(tn + idx16) * kD;

  v8f c0 = {}, c1 = {}, c2 = {};
#pragma unroll
  for (int kc = 0; kc < kD; kc += 32) {
    union { v16h v; v8h h[2]; } ah, al, bh, bl;
    ah.h[0] = *(const v8h*)(arh + kc + halfw * 8);
    ah.h[1] = *(const v8h*)(arh + kc + 16 + halfw * 8);
    al.h[0] = *(const v8h*)(arl + kc + halfw * 8);
    al.h[1] = *(const v8h*)(arl + kc + 16 + halfw * 8);
    bh.h[0] = *(const v8h*)(brh + kc + halfw * 8);
    bh.h[1] = *(const v8h*)(brh + kc + 16 + halfw * 8);
    bl.h[0] = *(const v8h*)(brl + kc + halfw * 8);
    bl.h[1] = *(const v8h*)(brl + kc + 16 + halfw * 8);

    c0 = __builtin_amdgcn_wmma_f32_16x16x32_f16(false, ah.v, false, bh.v,
                                                (short)0, c0, false, false);
    c1 = __builtin_amdgcn_wmma_f32_16x16x32_f16(false, ah.v, false, bl.v,
                                                (short)0, c1, false, false);
    c2 = __builtin_amdgcn_wmma_f32_16x16x32_f16(false, al.v, false, bh.v,
                                                (short)0, c2, false, false);
  }

  // C/D layout: VGPR r, lane (halfw, idx16) -> (m = r + 8*halfw, n = idx16).
#pragma unroll
  for (int r = 0; r < 8; ++r) {
    const int gi = tm + r + 8 * halfw;
    const int gj = tn + idx16;
    float v = 10.0f * ((c0[r] + c1[r] + c2[r]) - 1.0f);   // -(1-C)/0.1
    if (gi == gj) v = -10000.0f;                          // -1000/0.1
    logK[(size_t)gi * kN + gj] = v;
  }
}

// ---------------------------------------------------------------------------
// Online logsumexp helpers.
// ---------------------------------------------------------------------------
__device__ __forceinline__ void lse_push(float v, float& m, float& s) {
  if (v > m) { s = s * expf(m - v) + 1.0f; m = v; }
  else       { s += expf(v - m); }
}
__device__ __forceinline__ void lse_merge(float m2, float s2, float& m, float& s) {
  const float M = fmaxf(m, m2);
  s = s * expf(m - M) + s2 * expf(m2 - M);
  m = M;
}

// ---------------------------------------------------------------------------
// 3) log_u[i] = -logsumexp_j(logK[i,j] + log_v[j]).  One block per row.
// ---------------------------------------------------------------------------
__global__ void sot_row_lse(const float* __restrict__ logK,
                            const float* __restrict__ log_v,
                            float* __restrict__ log_u) {
  __shared__ float sm[256], ss[256];
  const int i = blockIdx.x;
  const float* row = logK + (size_t)i * kN;

  float m = -INFINITY, s = 0.0f;
  for (int j = threadIdx.x; j < kN; j += 256)
    lse_push(row[j] + log_v[j], m, s);

  sm[threadIdx.x] = m; ss[threadIdx.x] = s;
  __syncthreads();
  for (int st = 128; st > 0; st >>= 1) {
    if (threadIdx.x < st) {
      float m1 = sm[threadIdx.x], s1 = ss[threadIdx.x];
      lse_merge(sm[threadIdx.x + st], ss[threadIdx.x + st], m1, s1);
      sm[threadIdx.x] = m1; ss[threadIdx.x] = s1;
    }
    __syncthreads();
  }
  if (threadIdx.x == 0) log_u[i] = -(sm[0] + logf(ss[0]));
}

// ---------------------------------------------------------------------------
// 4a) Column LSE, stage 1: each block owns 256 columns x 512 rows, streams
//     coalesced rows, keeps per-column online (m,s) in registers.
// ---------------------------------------------------------------------------
__global__ void sot_col_partial(const float* __restrict__ logK,
                                const float* __restrict__ log_u,
                                float* __restrict__ pm,
                                float* __restrict__ ps) {
  const int j  = blockIdx.x * 256 + threadIdx.x;
  const int i0 = blockIdx.y * kRCH;
  float m = -INFINITY, s = 0.0f;
  for (int i = i0; i < i0 + kRCH; ++i)
    lse_push(logK[(size_t)i * kN + j] + log_u[i], m, s);
  pm[(size_t)blockIdx.y * kN + j] = m;
  ps[(size_t)blockIdx.y * kN + j] = s;
}

// 4b) Column LSE, stage 2: merge the kRC partials per column.
__global__ void sot_col_combine(const float* __restrict__ pm,
                                const float* __restrict__ ps,
                                float* __restrict__ log_v) {
  const int j = blockIdx.x * 256 + threadIdx.x;
  float m = -INFINITY, s = 0.0f;
#pragma unroll
  for (int c = 0; c < kRC; ++c)
    lse_merge(pm[(size_t)c * kN + j], ps[(size_t)c * kN + j], m, s);
  log_v[j] = -(m + logf(s));
}

// ---------------------------------------------------------------------------
// 5) max(sigmoid(log_P)) == sigmoid(max(log_P)) (monotone): max-reduce log_P.
// ---------------------------------------------------------------------------
__global__ void sot_max_partial(const float* __restrict__ logK,
                                const float* __restrict__ log_u,
                                const float* __restrict__ log_v,
                                float* __restrict__ rowmax) {
  __shared__ float sm[256];
  const int i = blockIdx.x;
  const float* row = logK + (size_t)i * kN;
  float m = -INFINITY;
  for (int j = threadIdx.x; j < kN; j += 256)
    m = fmaxf(m, row[j] + log_v[j]);
  sm[threadIdx.x] = m;
  __syncthreads();
  for (int st = 128; st > 0; st >>= 1) {
    if (threadIdx.x < st)
      sm[threadIdx.x] = fmaxf(sm[threadIdx.x], sm[threadIdx.x + st]);
    __syncthreads();
  }
  if (threadIdx.x == 0) rowmax[i] = sm[0] + log_u[i];
}

__global__ void sot_max_combine(const float* __restrict__ rowmax,
                                float* __restrict__ maxval) {
  __shared__ float sm[256];
  float m = -INFINITY;
  for (int i = threadIdx.x; i < kN; i += 256) m = fmaxf(m, rowmax[i]);
  sm[threadIdx.x] = m;
  __syncthreads();
  for (int st = 128; st > 0; st >>= 1) {
    if (threadIdx.x < st)
      sm[threadIdx.x] = fmaxf(sm[threadIdx.x], sm[threadIdx.x + st]);
    __syncthreads();
  }
  if (threadIdx.x == 0) maxval[0] = sm[0];
}

// ---------------------------------------------------------------------------
// 6) In-place finalize: out = sigmoid(log_P)/sigmoid(maxLogP), diag = 1.
//    sigmoid(p)/sigmoid(M) = (1+e^-M)/(1+e^-p).  float4 per thread.
// ---------------------------------------------------------------------------
__global__ void sot_final(float* __restrict__ out,
                          const float* __restrict__ log_u,
                          const float* __restrict__ log_v,
                          const float* __restrict__ maxval) {
  const size_t base = ((size_t)blockIdx.x * 256 + threadIdx.x) * 4;
  const int i = (int)(base >> 13);          // / 8192
  const int j = (int)(base & (kN - 1));     // % 8192 (row-aligned, N%4==0)
  const float num = 1.0f + expf(-maxval[0]);
  const float lu  = log_u[i];

  float4 k = *(const float4*)(out + base);
  float4 r;
  r.x = num / (1.0f + expf(-(lu + k.x + log_v[j + 0])));
  r.y = num / (1.0f + expf(-(lu + k.y + log_v[j + 1])));
  r.z = num / (1.0f + expf(-(lu + k.z + log_v[j + 2])));
  r.w = num / (1.0f + expf(-(lu + k.w + log_v[j + 3])));
  const int dj = i - j;
  if (dj == 0) r.x = 1.0f;
  if (dj == 1) r.y = 1.0f;
  if (dj == 2) r.z = 1.0f;
  if (dj == 3) r.w = 1.0f;
  *(float4*)(out + base) = r;
}

// ---------------------------------------------------------------------------
// Host-side orchestration (graph-capture safe: stream-only, deterministic).
// ---------------------------------------------------------------------------
extern "C" void kernel_launch(void* const* d_in, const int* in_sizes, int n_in,
                              void* d_out, int out_size, void* d_ws, size_t ws_size,
                              hipStream_t stream) {
  (void)in_sizes; (void)n_in; (void)out_size; (void)ws_size;

  const float* X = (const float*)d_in[0];
  float* out = (float*)d_out;               // doubles as the log_K buffer
  char* ws = (char*)d_ws;

  // Workspace layout (256B aligned), total ~5.4 MB.
  size_t off = 0;
  auto take = [&](size_t bytes) { size_t o = off; off += (bytes + 255) & ~(size_t)255; return o; };
  _Float16* XnHi  = (_Float16*)(ws + take((size_t)kN * kD * 2));
  _Float16* XnLo  = (_Float16*)(ws + take((size_t)kN * kD * 2));
  float* log_u    = (float*)(ws + take((size_t)kN * 4));
  float* log_v    = (float*)(ws + take((size_t)kN * 4));
  float* pm       = (float*)(ws + take((size_t)kRC * kN * 4));
  float* ps       = (float*)(ws + take((size_t)kRC * kN * 4));
  float* rowmax   = (float*)(ws + take((size_t)kN * 4));
  float* maxval   = (float*)(ws + take(256));

  // 1) normalize + split-f16
  sot_normalize<<<kN / 8, 256, 0, stream>>>(X, XnHi, XnLo);

  // 2) log_K via WMMA (write 256 MB once)
  sot_logk<<<dim3(kN / 16, kN / 128), 256, 0, stream>>>(XnHi, XnLo, out);

  // 3) Sinkhorn, 10 iterations (log_v starts at 0)
  hipMemsetAsync(log_v, 0, (size_t)kN * 4, stream);
  for (int it = 0; it < 10; ++it) {
    sot_row_lse<<<kN, 256, 0, stream>>>(out, log_v, log_u);
    sot_col_partial<<<dim3(kN / 256, kRC), 256, 0, stream>>>(out, log_u, pm, ps);
    sot_col_combine<<<kN / 256, 256, 0, stream>>>(pm, ps, log_v);
  }

  // 4) global max of log_P, then finalize in place
  sot_max_partial<<<kN, 256, 0, stream>>>(out, log_u, log_v, rowmax);
  sot_max_combine<<<1, 256, 0, stream>>>(rowmax, maxval);
  sot_final<<<(unsigned)((size_t)kN * kN / 4 / 256), 256, 0, stream>>>(out, log_u, log_v, maxval);
}
